// MultiAttentionHeadCompact_45019847196894
// MI455X (gfx1250) — matrix operand order repaired and checked by reading, not compiled, not run on hardware
//
#include <hip/hip_runtime.h>

// ---------------------------------------------------------------------------
// MHA forward on MI455X (gfx1250): all GEMMs on v_wmma_f32_16x16x32_bf16.
// GEMM: 64x64 register-blocked tile per wave (16 WMMA per 16 b128 loads).
// Attention: flash-style streaming, no NxN materialization, f32 accum.
// ---------------------------------------------------------------------------

typedef __attribute__((ext_vector_type(16))) __bf16   v16bf;
typedef __attribute__((ext_vector_type(8)))  float    v8f;
typedef __attribute__((ext_vector_type(4)))  unsigned int u32x4;

union Frag {               // one WMMA A/B operand: 8 VGPRs / 32 bytes per lane
  v16bf v;
  u32x4 q[2];
  unsigned short u[16];
};

__device__ inline unsigned short f2bf(float f) {   // round-to-nearest-even
  unsigned int x = __float_as_uint(f);
  x += 0x7FFFu + ((x >> 16) & 1u);
  return (unsigned short)(x >> 16);
}

__device__ inline v8f v8f_zero() {
  v8f z = {0.f, 0.f, 0.f, 0.f, 0.f, 0.f, 0.f, 0.f};
  return z;
}

static constexpr int Bsz = 4, Nseq = 2048, Hh = 16, Dh = 64;
static constexpr int Mrows = Bsz * Nseq;   // 8192
static constexpr int Kdim  = 1024;         // model dim = H*Dh = 1024 everywhere
static constexpr int Nfeat = 1024;

// ---------------- conversion kernels ----------------
__global__ void k_cvt(const float* __restrict__ in, unsigned short* __restrict__ out, int n) {
  for (int i = blockIdx.x * blockDim.x + threadIdx.x; i < n; i += gridDim.x * blockDim.x)
    out[i] = f2bf(in[i]);
}

// f32 [rows][cols] -> bf16 [cols][rows] (weights pre-transposed so WMMA
// B-fragments become contiguous per-lane 32B loads)
__global__ void k_cvt_t(const float* __restrict__ in, unsigned short* __restrict__ out,
                        int rows, int cols) {
  int n = rows * cols;
  for (int i = blockIdx.x * blockDim.x + threadIdx.x; i < n; i += gridDim.x * blockDim.x) {
    int r = i / cols, c = i - r * cols;
    out[c * rows + r] = f2bf(in[i]);
  }
}

// ---------------- WMMA GEMM: C[8192x1024] = A[8192x1024] * Bt^T ----------------
// One wave computes a 64x64 tile = 4x4 grid of 16x16 WMMA tiles.
// Per K-step of 32: 4 A-frags + 4 B-frags (16 b128 loads) drive 16 WMMAs.
// mode 0: bf16 out, head-split  [b,h,n,64]   (for Q and K)
// mode 1: bf16 out, transposed  [b,h,64,n]   (for V; vectorized b128 store)
// mode 2: f32 out, row-major    [8192,1024]  (final @ W_o -> d_out)
__global__ void __launch_bounds__(256) k_gemm(const unsigned short* __restrict__ A,
                                              const unsigned short* __restrict__ Bt,
                                              void* __restrict__ dst, int mode) {
  const int wave = threadIdx.x >> 5;
  const int lane = threadIdx.x & 31;
  const int id = blockIdx.x * 8 + wave;
  const int nt = id & (Nfeat / 64 - 1);          // 16 n-tiles; waves share mt
  const int mt = id >> 4;                        // 128 m-tiles
  const int m0 = mt * 64, n0 = nt * 64;
  const int lr = lane & 15, hi = lane >> 4;

  const unsigned short* arow0 = A  + (size_t)(m0 + lr) * Kdim + (hi << 3); // runs of 8
  const unsigned short* brow0 = Bt + (size_t)(n0 + lr) * Kdim + (hi << 3); // runs of 8

  v8f acc[4][4];
#pragma unroll
  for (int ms = 0; ms < 4; ++ms)
#pragma unroll
    for (int ns = 0; ns < 4; ++ns) acc[ms][ns] = v8f_zero();

  for (int k0 = 0; k0 < Kdim; k0 += 32) {
    Frag a[4];
#pragma unroll
    for (int ms = 0; ms < 4; ++ms) {
      const unsigned short* ap = arow0 + (size_t)(ms * 16) * Kdim + k0;
      a[ms].q[0] = *(const u32x4*)ap;
      a[ms].q[1] = *(const u32x4*)(ap + 16);
    }
#pragma unroll
    for (int ns = 0; ns < 4; ++ns) {
      Frag b;
      const unsigned short* bp = brow0 + (size_t)(ns * 16) * Kdim + k0;
      b.q[0] = *(const u32x4*)bp;
      b.q[1] = *(const u32x4*)(bp + 16);
#pragma unroll
      for (int ms = 0; ms < 4; ++ms)
        acc[ms][ns] = __builtin_amdgcn_wmma_f32_16x16x32_bf16(
            false, a[ms].v, false, b.v, (short)0, acc[ms][ns], false, false);
    }
  }

  // ---- epilogue: 16 sub-tiles ----
#pragma unroll
  for (int ms = 0; ms < 4; ++ms) {
#pragma unroll
    for (int ns = 0; ns < 4; ++ns) {
      const v8f c = acc[ms][ns];
      const int cn = n0 + ns * 16 + lr;          // column (feature)
      const int mbase = m0 + ms * 16 + (hi << 3);
      if (mode == 0) {                           // Q/K: [b,h,n,64] bf16
        unsigned short* Q = (unsigned short*)dst;
        int h = cn >> 6, dq = cn & 63;
#pragma unroll
        for (int r = 0; r < 8; ++r) {
          int m = mbase + r;
          int b = m >> 11, nn = m & (Nseq - 1);
          Q[((size_t)((b * Hh + h) * Nseq + nn) << 6) + dq] = f2bf(c[r]);
        }
      } else if (mode == 1) {                    // V transposed: [b,h,64,n] bf16
        unsigned short* V = (unsigned short*)dst;
        int h = cn >> 6, dq = cn & 63;
        int b = mbase >> 11, nn0 = mbase & (Nseq - 1);
        union { u32x4 q; unsigned short u[8]; } pk;
#pragma unroll
        for (int r = 0; r < 8; ++r) pk.u[r] = f2bf(c[r]);
        *(u32x4*)(V + (size_t)((b * Hh + h) * Dh + dq) * Nseq + nn0) = pk.q;
      } else {                                   // final output, f32 row-major
        float* O = (float*)dst;
#pragma unroll
        for (int r = 0; r < 8; ++r)
          O[(size_t)(mbase + r) * Nfeat + cn] = c[r];
      }
    }
  }
}

// ---------------- flash attention ----------------
// One wave per 16-row q-tile per (b,h). Streams causal key blocks of 32:
// S(16x32) via 4 WMMAs, online softmax (row = 16 lanes of one half-wave),
// P -> bf16 via per-wave LDS re-layout, O(16x64) += P@V via 4 WMMAs.
__global__ void __launch_bounds__(256) k_flash(const unsigned short* __restrict__ Q,
                                               const unsigned short* __restrict__ Kc,
                                               const unsigned short* __restrict__ Vt,
                                               unsigned short* __restrict__ O) {
  __shared__ __align__(16) unsigned short lds[8 * 16 * 32];   // 1KB per wave
  const int wave = threadIdx.x >> 5;
  const int lane = threadIdx.x & 31;
  const int id = blockIdx.x * 8 + wave;
  const int qt = id & 127;                 // 128 q-tiles per (b,h)
  const int bh = id >> 7;                  // 0..63
  const int qbase = qt * 16;
  const int lr = lane & 15, hi = lane >> 4;

  const unsigned short* Qh = Q  + (size_t)bh * Nseq * Dh;
  const unsigned short* Kh = Kc + (size_t)bh * Nseq * Dh;
  const unsigned short* Vh = Vt + (size_t)bh * Dh * Nseq;
  unsigned short* pbuf = lds + wave * (16 * 32);

  // Q A-fragments: 16x64 as two 16x32 chunks, loaded once
  Frag aq[2];
  {
    const unsigned short* qrow = Qh + (size_t)(qbase + lr) * Dh + (hi << 3);
#pragma unroll
    for (int ch = 0; ch < 2; ++ch) {
      aq[ch].q[0] = *(const u32x4*)(qrow + ch * 32);
      aq[ch].q[1] = *(const u32x4*)(qrow + ch * 32 + 16);
    }
  }

  float mrow[8], lrow[8];
  v8f o[4];
#pragma unroll
  for (int r = 0; r < 8; ++r) { mrow[r] = -3.0e38f; lrow[r] = 0.f; }
#pragma unroll
  for (int t = 0; t < 4; ++t) o[t] = v8f_zero();

  const int kend = qbase + 16;                      // causal: keys <= qbase+15
  for (int kb = 0; kb < kend; kb += 32) {
    // ---- S = (Q K^T) for 32 keys: two 16x16 C tiles ----
    v8f sc[2];
#pragma unroll
    for (int s = 0; s < 2; ++s) {
      sc[s] = v8f_zero();
      const unsigned short* krow = Kh + (size_t)(kb + 16 * s + lr) * Dh + (hi << 3);
#pragma unroll
      for (int ch = 0; ch < 2; ++ch) {
        Frag bk;
        bk.q[0] = *(const u32x4*)(krow + ch * 32);
        bk.q[1] = *(const u32x4*)(krow + ch * 32 + 16);
        sc[s] = __builtin_amdgcn_wmma_f32_16x16x32_bf16(false, aq[ch].v, false, bk.v,
                                                        (short)0, sc[s], false, false);
      }
    }
    // ---- scale + causal mask ----
#pragma unroll
    for (int s = 0; s < 2; ++s) {
      int kg = kb + 16 * s + lr;
#pragma unroll
      for (int r = 0; r < 8; ++r) {
        int qg = qbase + (hi << 3) + r;
        float v = sc[s][r] * 0.125f;                // 1/sqrt(64)
        sc[s][r] = (kg > qg) ? -1.0e9f : v;
      }
    }
    // ---- online softmax: row = one VGPR across a 16-lane half-wave ----
#pragma unroll
    for (int r = 0; r < 8; ++r) {
      float bm = fmaxf(sc[0][r], sc[1][r]);
      bm = fmaxf(bm, __shfl_xor(bm, 1));
      bm = fmaxf(bm, __shfl_xor(bm, 2));
      bm = fmaxf(bm, __shfl_xor(bm, 4));
      bm = fmaxf(bm, __shfl_xor(bm, 8));
      float mnew  = fmaxf(mrow[r], bm);
      float alpha = __expf(mrow[r] - mnew);
      float p0 = __expf(sc[0][r] - mnew);
      float p1 = __expf(sc[1][r] - mnew);
      sc[0][r] = p0; sc[1][r] = p1;
      float rs = p0 + p1;
      rs += __shfl_xor(rs, 1);
      rs += __shfl_xor(rs, 2);
      rs += __shfl_xor(rs, 4);
      rs += __shfl_xor(rs, 8);
      lrow[r] = lrow[r] * alpha + rs;
      mrow[r] = mnew;
#pragma unroll
      for (int t = 0; t < 4; ++t) o[t][r] *= alpha;
    }
    // ---- P: C-layout f32 -> bf16 row-major in LDS -> A-fragment ----
#pragma unroll
    for (int s = 0; s < 2; ++s)
#pragma unroll
      for (int r = 0; r < 8; ++r)
        pbuf[((hi << 3) + r) * 32 + 16 * s + lr] = f2bf(sc[s][r]);
    asm volatile("s_wait_dscnt 0x0" ::: "memory");   // wave-synchronous re-layout
    Frag pa;
    pa.q[0] = *(const u32x4*)(pbuf + lr * 32 + (hi << 3));
    pa.q[1] = *(const u32x4*)(pbuf + lr * 32 + 16 + (hi << 3));
    // ---- O += P @ V (V pre-transposed -> contiguous B-fragment loads) ----
#pragma unroll
    for (int t = 0; t < 4; ++t) {
      const unsigned short* vrow = Vh + (size_t)(t * 16 + lr) * Nseq + kb + (hi << 3);
      Frag bv;
      bv.q[0] = *(const u32x4*)vrow;
      bv.q[1] = *(const u32x4*)(vrow + 16);
      o[t] = __builtin_amdgcn_wmma_f32_16x16x32_bf16(false, pa.v, false, bv.v,
                                                     (short)0, o[t], false, false);
    }
  }

  // ---- normalize and store merged [b*N+n][H*64] bf16 ----
  const int b = bh >> 4, h = bh & 15;
#pragma unroll
  for (int t = 0; t < 4; ++t) {
    int col = h * 64 + t * 16 + lr;
#pragma unroll
    for (int r = 0; r < 8; ++r) {
      int tok = b * Nseq + qbase + (hi << 3) + r;
      O[(size_t)tok * 1024 + col] = f2bf(o[t][r] / lrow[r]);
    }
  }
}

// ---------------- launch ----------------
extern "C" void kernel_launch(void* const* d_in, const int* in_sizes, int n_in,
                              void* d_out, int out_size, void* d_ws, size_t ws_size,
                              hipStream_t stream) {
  const float* enc = (const float*)d_in[0];
  // d_in[1] = causal mask; applied analytically (kg > qg) in k_flash
  const float* Wq = (const float*)d_in[2];
  const float* Wk = (const float*)d_in[3];
  const float* Wv = (const float*)d_in[4];
  const float* Wo = (const float*)d_in[5];

  char* ws = (char*)d_ws;                     // ~92 MB total
  unsigned short* Xb  = (unsigned short*)ws;                       // 16.8 MB
  unsigned short* Wqt = Xb  + (size_t)Mrows * Kdim;                // 2 MB each
  unsigned short* Wkt = Wqt + (size_t)Kdim * Nfeat;
  unsigned short* Wvt = Wkt + (size_t)Kdim * Nfeat;
  unsigned short* Wot = Wvt + (size_t)Kdim * Nfeat;
  unsigned short* Qb  = Wot + (size_t)Kdim * Nfeat;                // 16.8 MB each
  unsigned short* Kb  = Qb  + (size_t)Mrows * Kdim;
  unsigned short* Vtb = Kb  + (size_t)Mrows * Kdim;
  unsigned short* Ob  = Vtb + (size_t)Mrows * Kdim;

  k_cvt  <<<4096, 256, 0, stream>>>(enc, Xb, Mrows * Kdim);
  k_cvt_t<<<1024, 256, 0, stream>>>(Wq, Wqt, Kdim, Nfeat);
  k_cvt_t<<<1024, 256, 0, stream>>>(Wk, Wkt, Kdim, Nfeat);
  k_cvt_t<<<1024, 256, 0, stream>>>(Wv, Wvt, Kdim, Nfeat);
  k_cvt_t<<<1024, 256, 0, stream>>>(Wo, Wot, Kdim, Nfeat);

  // 2048 waves per GEMM: 128 m-tiles x 16 n-tiles of 64x64
  k_gemm<<<256, 256, 0, stream>>>(Xb, Wqt, Qb,   0);   // Q  [b,h,n,64]
  k_gemm<<<256, 256, 0, stream>>>(Xb, Wkt, Kb,   0);   // K  [b,h,n,64]
  k_gemm<<<256, 256, 0, stream>>>(Xb, Wvt, Vtb,  1);   // V^T[b,h,64,n]
  k_flash<<<1024, 256, 0, stream>>>(Qb, Kb, Vtb, Ob);  // O  [b*n, 1024] bf16
  k_gemm<<<256, 256, 0, stream>>>(Ob, Wot, d_out, 2);  // out f32 [b*n, 1024]
}
